// EdgeAttrPredictor_29360396436025
// MI455X (gfx1250) — matrix operand order repaired and checked by reading, not compiled, not run on hardware
//
#include <hip/hip_runtime.h>

typedef __attribute__((ext_vector_type(16))) _Float16 v16h;
typedef __attribute__((ext_vector_type(2)))  _Float16 h2v;
typedef __attribute__((ext_vector_type(8)))  float    v8f;

#define LATENT 64
#define HID 32
#define EDIM 4

union FragB { v16h h; unsigned u[8]; };
union U2 { h2v h; unsigned u; };

// pack two f32 -> one dword of two f16 (v_cvt_pk_f16_f32)
__device__ __forceinline__ unsigned cvt2(float a, float b) {
  union { _Float16 h[2]; unsigned u; } t;
  t.h[0] = (_Float16)a; t.h[1] = (_Float16)b;
  return t.u;
}

// single-instruction relu: med3(x, 0, +inf) == clamp(x, 0, inf)
__device__ __forceinline__ float relu1(float x) {
  return __builtin_amdgcn_fmed3f(x, 0.0f, __builtin_inff());
}

// A-fragment (16-bit, 16xK tile): lane holds row m; K contiguous as
// {half*16 + 0..15} within each 32-deep K block.
__device__ __forceinline__ v16h load_afrag(const _Float16* rowbase, int hf) {
  union { v16h h; uint4 u[2]; } fr;
  fr.u[0] = *(const uint4*)(rowbase + hf * 16);
  fr.u[1] = *(const uint4*)(rowbase + hf * 16 + 8);
  return fr.h;
}

// C-fragment bias: element v <-> m = off + v (off = mb*16 + half*8)
__device__ __forceinline__ v8f load_biasfrag(const float* b, int off) {
  float4 x0 = *(const float4*)(b + off);
  float4 x1 = *(const float4*)(b + off + 4);
  v8f c;
  c[0] = x0.x; c[1] = x0.y; c[2] = x0.z; c[3] = x0.w;
  c[4] = x1.x; c[5] = x1.y; c[6] = x1.z; c[7] = x1.w;
  return c;
}

#define WMMA(A, B, C) __builtin_amdgcn_wmma_f32_16x16x32_f16(false, (A), false, (B), (short)0, (C), false, false)

__global__ void __launch_bounds__(256) edge_mlp_kernel(
    const float* __restrict__ x, const int* __restrict__ eidx,
    const float* __restrict__ Wd, const float* __restrict__ bd,
    const float* __restrict__ W1, const float* __restrict__ b1,
    const float* __restrict__ W2, const float* __restrict__ b2,
    const float* __restrict__ W3, const float* __restrict__ b3,
    float* __restrict__ out, int E, int NT)
{
  // Transposed fp16 weight copies (rows = output unit m, cols = input k)
  __shared__ __align__(16) _Float16 sW1[HID * 128];  // [m][k] = W1[k][m]
  __shared__ __align__(16) _Float16 sW2[HID * HID];  // [m][k] = W2[k][m]
  __shared__ __align__(16) _Float16 sW3[16 * HID];   // padded: m<4 ? W3[k][m] : 0
  __shared__ __align__(16) _Float16 sWd[16 * 128];   // padded: m<4 ? Wd[k][m] : 0
  __shared__ __align__(16) float sB1[HID];
  __shared__ __align__(16) float sB2[HID];
  __shared__ __align__(16) float sB3[16];            // fused b3 + b_direct, padded

  const int tid = threadIdx.x;
  for (int i = tid; i < HID * 128; i += 256) { int m = i >> 7, k = i & 127; sW1[i] = (_Float16)W1[k * HID + m]; }
  for (int i = tid; i < HID * HID; i += 256) { int m = i >> 5, k = i & 31;  sW2[i] = (_Float16)W2[k * HID + m]; }
  for (int i = tid; i < 16 * HID; i += 256)  { int m = i >> 5, k = i & 31;  sW3[i] = (m < EDIM) ? (_Float16)W3[k * EDIM + m] : (_Float16)0.f; }
  for (int i = tid; i < 16 * 128; i += 256)  { int m = i >> 7, k = i & 127; sWd[i] = (m < EDIM) ? (_Float16)Wd[k * EDIM + m] : (_Float16)0.f; }
  for (int i = tid; i < HID; i += 256) { sB1[i] = b1[i]; sB2[i] = b2[i]; }
  for (int i = tid; i < 16; i += 256)  { sB3[i] = (i < EDIM) ? (b3[i] + bd[i]) : 0.f; }
  __syncthreads();

  const int lane = tid & 31;
  const int hf = lane >> 4;       // lane half (0: lanes 0-15, 1: lanes 16-31)
  const int el = lane & 15;       // edge slot within tile / matrix row index
  const int wave = blockIdx.x * (blockDim.x >> 5) + (tid >> 5);
  const int nWaves = gridDim.x * (blockDim.x >> 5);

  for (int tile = wave; tile < NT; tile += nWaves) {
    int e = tile * 16 + el;
    int ec = (e < E) ? e : (E - 1);
    int s = eidx[ec];
    int t = eidx[E + ec];
    const float4* ps = (const float4*)(x + (long)s * LATENT);
    const float4* pt = (const float4*)(x + (long)t * LATENT);

    // B fragments of x_cat^T (128x16): bx[0..1] = min (k 0..63), bx[2..3] = max.
    // Convert to fp16 first (monotonic), then packed fp16 min/max.
    v16h bx[4];
#pragma unroll
    for (int g = 0; g < 2; ++g) {
      int c4 = g * 8 + hf * 4;   // float4 index of column (g*32 + hf*16)
      FragB fmn, fmx;
#pragma unroll
      for (int p = 0; p < 4; ++p) {
        float4 s4 = ps[c4 + p];
        float4 t4 = pt[c4 + p];
        U2 s0, s1, t0, t1, m;
        s0.u = cvt2(s4.x, s4.y); s1.u = cvt2(s4.z, s4.w);
        t0.u = cvt2(t4.x, t4.y); t1.u = cvt2(t4.z, t4.w);
        m.h = __builtin_elementwise_min(s0.h, t0.h); fmn.u[2 * p]     = m.u;
        m.h = __builtin_elementwise_min(s1.h, t1.h); fmn.u[2 * p + 1] = m.u;
        m.h = __builtin_elementwise_max(s0.h, t0.h); fmx.u[2 * p]     = m.u;
        m.h = __builtin_elementwise_max(s1.h, t1.h); fmx.u[2 * p + 1] = m.u;
      }
      bx[g] = fmn.h; bx[g + 2] = fmx.h;
    }

    // ---- layer 1: h1^T = relu(W1^T @ xcat^T + b1)  (M=32 -> two C frags) ----
    v8f acc0 = load_biasfrag(sB1, hf * 8);
    v8f acc1 = load_biasfrag(sB1, 16 + hf * 8);
#pragma unroll
    for (int kb = 0; kb < 4; ++kb) {
      v16h a0 = load_afrag(sW1 + el * 128 + kb * 32, hf);
      v16h a1 = load_afrag(sW1 + (16 + el) * 128 + kb * 32, hf);
      acc0 = WMMA(a0, bx[kb], acc0);
      acc1 = WMMA(a1, bx[kb], acc1);
    }
#pragma unroll
    for (int v = 0; v < 8; ++v) { acc0[v] = relu1(acc0[v]); acc1[v] = relu1(acc1[v]); }

    // C -> B relayout for h1 (fp16 pack + lane-half swap)
    FragB bh1;
#pragma unroll
    for (int v = 0; v < 4; ++v) {
      unsigned p0 = cvt2(acc0[2 * v], acc0[2 * v + 1]);
      unsigned p1 = cvt2(acc1[2 * v], acc1[2 * v + 1]);
      unsigned q0 = __shfl_xor(p0, 16, 32);
      unsigned q1 = __shfl_xor(p1, 16, 32);
      bh1.u[v]     = hf ? q1 : p0;
      bh1.u[v + 4] = hf ? p1 : q0;
    }

    // ---- layer 2 (residual): h2^T = relu(W2^T @ h1^T + b2 + h1^T) ----
    v8f acc20 = acc0 + load_biasfrag(sB2, hf * 8);
    v8f acc21 = acc1 + load_biasfrag(sB2, 16 + hf * 8);
    {
      v16h a20 = load_afrag(sW2 + el * HID, hf);
      v16h a21 = load_afrag(sW2 + (16 + el) * HID, hf);
      acc20 = WMMA(a20, bh1.h, acc20);
      acc21 = WMMA(a21, bh1.h, acc21);
    }
#pragma unroll
    for (int v = 0; v < 8; ++v) { acc20[v] = relu1(acc20[v]); acc21[v] = relu1(acc21[v]); }

    FragB bh2;
#pragma unroll
    for (int v = 0; v < 4; ++v) {
      unsigned p0 = cvt2(acc20[2 * v], acc20[2 * v + 1]);
      unsigned p1 = cvt2(acc21[2 * v], acc21[2 * v + 1]);
      unsigned q0 = __shfl_xor(p0, 16, 32);
      unsigned q1 = __shfl_xor(p1, 16, 32);
      bh2.u[v]     = hf ? q1 : p0;
      bh2.u[v + 4] = hf ? p1 : q0;
    }

    // ---- layer 3 + direct path: out^T = W3^T @ h2^T + Wd^T @ xcat^T + (b3+bd) ----
    v8f oacc = load_biasfrag(sB3, hf * 8);
    {
      v16h a3 = load_afrag(sW3 + el * HID, hf);
      oacc = WMMA(a3, bh2.h, oacc);
    }
#pragma unroll
    for (int kb = 0; kb < 4; ++kb) {
      v16h ad = load_afrag(sWd + el * 128 + kb * 32, hf);
      oacc = WMMA(ad, bx[kb], oacc);
    }

    // C frag: lane el (half 0) holds m = 0..7 -> outputs 0..3 of edge e
    if (hf == 0 && e < E) {
      float4 r;
      r.x = oacc[0]; r.y = oacc[1]; r.z = oacc[2]; r.w = oacc[3];
      *(float4*)(out + (long)e * EDIM) = r;
    }
  }
}

extern "C" void kernel_launch(void* const* d_in, const int* in_sizes, int n_in,
                              void* d_out, int out_size, void* d_ws, size_t ws_size,
                              hipStream_t stream) {
  const float* x  = (const float*)d_in[0];
  const int*   ei = (const int*)d_in[1];
  const float* Wd = (const float*)d_in[2];
  const float* bd = (const float*)d_in[3];
  const float* W1 = (const float*)d_in[4];
  const float* b1 = (const float*)d_in[5];
  const float* W2 = (const float*)d_in[6];
  const float* b2 = (const float*)d_in[7];
  const float* W3 = (const float*)d_in[8];
  const float* b3 = (const float*)d_in[9];
  float* out = (float*)d_out;

  int E  = in_sizes[1] / 2;
  int NT = (E + 15) / 16;
  int blocks = 1024;                       // 8 waves/block
  int need = (NT + 7) / 8;
  if (blocks > need) blocks = (need > 0) ? need : 1;

  edge_mlp_kernel<<<blocks, 256, 0, stream>>>(x, ei, Wd, bd, W1, b1, W2, b2, W3, b3, out, E, NT);
}
